// RCNNnet_4681514353324
// MI455X (gfx1250) — matrix operand order repaired and checked by reading, not compiled, not run on hardware
//
#include <hip/hip_runtime.h>
#include <cstdint>
#include <cstddef>

// ---------------- problem constants ----------------
constexpr int B_    = 4;
constexpr int A_    = 70400;
constexpr int FEAT_ = 27648;
constexpr int NPRE  = 1024;
constexpr int NPOST = 512;
constexpr int HID   = 256;
constexpr int MROWS = B_ * NPOST;   // 2048
#define NMS_TH 0.7f

typedef _Float16 h16;
typedef _Float16 v16h __attribute__((ext_vector_type(16)));
typedef _Float16 v4h  __attribute__((ext_vector_type(4)));
typedef float    v8f  __attribute__((ext_vector_type(8)));

// monotone sortable key for float
__device__ __forceinline__ unsigned fkey(float f) {
  unsigned u = __float_as_uint(f);
  return (u & 0x80000000u) ? ~u : (u | 0x80000000u);
}

// ---------------- 1) per-anchor score = max over 3 classes ----------------
__global__ void score_kernel(const float* __restrict__ cls,
                             float* __restrict__ score, int n) {
  int i = blockIdx.x * blockDim.x + threadIdx.x;
  if (i < n) {
    float c0 = cls[i * 3 + 0], c1 = cls[i * 3 + 1], c2 = cls[i * 3 + 2];
    score[i] = fmaxf(c0, fmaxf(c1, c2));
  }
}

// ---------------- 2) top-1024 per batch: radix-select + bitonic ----------------
__global__ __launch_bounds__(1024) void topk_kernel(
    const float* __restrict__ score, const float* __restrict__ boxes,
    float* __restrict__ topbox, float* __restrict__ topscore) {
  __shared__ unsigned h[1024];
  __shared__ unsigned long long keys[2048];
  __shared__ unsigned cnt;
  __shared__ int Tbin;
  const int tid = threadIdx.x;
  const int b = blockIdx.x;
  const float* sc = score + (size_t)b * A_;

  h[tid] = 0u;
  if (tid == 0) cnt = 0u;
  __syncthreads();
  for (int i = tid; i < A_; i += 1024)
    atomicAdd(&h[fkey(sc[i]) >> 22], 1u);
  __syncthreads();
  // suffix scan (count of elements with bin >= tid)
  for (int s = 1; s < 1024; s <<= 1) {
    unsigned v = (tid + s < 1024) ? h[tid + s] : 0u;
    __syncthreads();
    h[tid] += v;
    __syncthreads();
  }
  if (h[tid] >= (unsigned)NPRE && (tid == 1023 || h[tid + 1] < (unsigned)NPRE))
    Tbin = tid;
  __syncthreads();
  const unsigned T = (unsigned)Tbin;
  for (int i = tid; i < A_; i += 1024) {
    unsigned u = fkey(sc[i]);
    if ((u >> 22) >= T) {
      unsigned p = atomicAdd(&cnt, 1u);
      if (p < 2048u)
        keys[p] = ((unsigned long long)u << 32) |
                  (unsigned long long)(0xFFFFFFFFu - (unsigned)i);
    }
  }
  __syncthreads();
  unsigned total = cnt; if (total > 2048u) total = 2048u;
  for (int p = tid; p < 2048; p += 1024)
    if ((unsigned)p >= total) keys[p] = 0ull;
  __syncthreads();
  // bitonic sort, descending
  for (int k = 2; k <= 2048; k <<= 1) {
    for (int j = k >> 1; j > 0; j >>= 1) {
      #pragma unroll
      for (int e = 0; e < 2; ++e) {
        int i = tid + e * 1024;
        int ix = i ^ j;
        if (ix > i) {
          unsigned long long a = keys[i], c = keys[ix];
          bool asc = (i & k) != 0;
          bool sw = asc ? (a > c) : (a < c);
          if (sw) { keys[i] = c; keys[ix] = a; }
        }
      }
      __syncthreads();
    }
  }
  // emit top 1024 (sorted descending; ties -> smaller index first)
  unsigned long long key = keys[tid];
  unsigned idx = 0xFFFFFFFFu - (unsigned)(key & 0xFFFFFFFFull);
  size_t dst = (size_t)b * NPRE + tid;
  topscore[dst] = sc[idx];
  const float* bx = boxes + ((size_t)b * A_ + idx) * 7;
  float* tb = topbox + dst * 7;
  #pragma unroll
  for (int d = 0; d < 7; ++d) tb[d] = bx[d];
}

// ---------------- 3) greedy BEV NMS + compaction to 512 rois ----------------
__global__ __launch_bounds__(1024) void nms_select_kernel(
    const float* __restrict__ topbox, const float* __restrict__ topscore,
    float* __restrict__ rois, float* __restrict__ roiscore) {
  __shared__ float sx1[NPRE], sy1[NPRE], sx2[NPRE], sy2[NPRE], sar[NPRE];
  __shared__ unsigned keep[NPRE];
  __shared__ unsigned pf[NPRE];
  const int tid = threadIdx.x;
  const int b = blockIdx.x;
  const float* tb = topbox + ((size_t)b * NPRE + tid) * 7;
  float x = tb[0], y = tb[1], dx = tb[3], dy = tb[4];
  sx1[tid] = x - 0.5f * dx; sx2[tid] = x + 0.5f * dx;
  sy1[tid] = y - 0.5f * dy; sy2[tid] = y + 0.5f * dy;
  sar[tid] = dx * dy;
  keep[tid] = 1u;
  __syncthreads();
  for (int i = 0; i < NPRE - 1; ++i) {
    if (keep[i] && tid > i) {
      float iw = fmaxf(fminf(sx2[i], sx2[tid]) - fmaxf(sx1[i], sx1[tid]), 0.f);
      float ih = fmaxf(fminf(sy2[i], sy2[tid]) - fmaxf(sy1[i], sy1[tid]), 0.f);
      float inter = iw * ih;
      float iou = inter / fmaxf(sar[i] + sar[tid] - inter, 1e-6f);
      if (iou > NMS_TH) keep[tid] = 0u;
    }
    __syncthreads();
  }
  unsigned kv = keep[tid];
  pf[tid] = kv;
  __syncthreads();
  for (int s = 1; s < 1024; s <<= 1) {
    unsigned v = (tid >= s) ? pf[tid - s] : 0u;
    __syncthreads();
    pf[tid] += v;
    __syncthreads();
  }
  unsigned incl = pf[tid];
  unsigned numkept = pf[NPRE - 1];
  unsigned excl = incl - kv;
  unsigned pos = kv ? excl : (numkept + ((unsigned)tid - excl));
  if (pos < (unsigned)NPOST) {
    size_t dst = (size_t)b * NPOST + pos;
    float* ro = rois + dst * 7;
    if (kv) {
      #pragma unroll
      for (int d = 0; d < 7; ++d) ro[d] = tb[d];
      roiscore[dst] = topscore[(size_t)b * NPRE + tid];
    } else {
      #pragma unroll
      for (int d = 0; d < 7; ++d) ro[d] = 0.f;
      roiscore[dst] = 0.f;
    }
  }
}

// ---------------- 4) weight fp32 -> fp16 ----------------
__global__ void f32_to_f16_kernel(const float* __restrict__ src,
                                  h16* __restrict__ dst, int n) {
  for (int i = blockIdx.x * blockDim.x + threadIdx.x; i < n;
       i += gridDim.x * blockDim.x)
    dst[i] = (h16)src[i];
}

// ---------------- 5) WMMA GEMM + fused bn_relu, N = 256 ----------------
// 256 threads = 8 waves (2x4), block tile 32 rows x 256 cols, KT = 32.
template <bool A_F32>
__global__ __launch_bounds__(256) void gemm_bnrelu_kernel(
    const void* __restrict__ Aptr,        // [M][K] f32 or f16
    const h16* __restrict__ W,            // [256][K] row-major (out,in)
    const float* __restrict__ g, const float* __restrict__ bias,
    h16* __restrict__ out,                // [M][256] f16 activations
    int K) {
  __shared__ __align__(32) h16 sA[2][32][16];  // [wave-row tile][lane][half]
  const int tid = threadIdx.x;
  const int lane = tid & 31;
  const int wave = tid >> 5;
  const int wr = wave >> 2;          // wave row 0..1
  const int wc = wave & 3;           // wave col 0..3
  const int rowbase = blockIdx.x * 32 + wr * 16;
  const int l15 = lane & 15;
  const int hi = lane >> 4;          // 0 or 1

  v8f acc[4];
  #pragma unroll
  for (int nt = 0; nt < 4; ++nt)
    #pragma unroll
    for (int e = 0; e < 8; ++e) acc[nt][e] = 0.f;

  const int sr = tid >> 3;           // staging row 0..31
  const int sk = (tid & 7) << 2;     // staging k 0,4,...,28
  const int kiters = K >> 5;

  for (int kb = 0; kb < kiters; ++kb) {
    const int k0 = kb << 5;
    const size_t base = (size_t)(blockIdx.x * 32 + sr) * K + (k0 + sk);
    float av[4];
    if (A_F32) {
      const float4 t = *reinterpret_cast<const float4*>((const float*)Aptr + base);
      av[0] = t.x; av[1] = t.y; av[2] = t.z; av[3] = t.w;
    } else {
      const v4h t = *reinterpret_cast<const v4h*>((const h16*)Aptr + base);
      av[0] = (float)t[0]; av[1] = (float)t[1];
      av[2] = (float)t[2]; av[3] = (float)t[3];
    }
    if (kb + 1 < kiters) {  // global_prefetch_b8 of next A tile
      if (A_F32) __builtin_prefetch((const float*)Aptr + base + 32, 0, 0);
      else       __builtin_prefetch((const h16*)Aptr + base + 32, 0, 0);
    }
    __syncthreads();
    // swizzled store so each lane's WMMA A-fragment is one contiguous 32B read
    #pragma unroll
    for (int q = 0; q < 4; ++q) {
      int kk = sk + q;
      int ln = (sr & 15) + ((kk >= 16) ? 16 : 0);
      int ix = kk & 15;
      sA[sr >> 4][ln][ix] = (h16)av[q];
    }
    __syncthreads();
    const v16h af = *reinterpret_cast<const v16h*>(&sA[wr][lane][0]);
    #pragma unroll
    for (int nt = 0; nt < 4; ++nt) {
      const int col = wc * 64 + nt * 16 + l15;
      const v16h bf = *reinterpret_cast<const v16h*>(
          W + (size_t)col * K + (size_t)(k0 + hi * 16));
      acc[nt] = __builtin_amdgcn_wmma_f32_16x16x32_f16(
          false, af, false, bf, (short)0, acc[nt], false, false);
    }
  }
  // fused bn_relu epilogue: relu(acc * g/sqrt(1+eps) + b) -> f16
  const float rs = rsqrtf(1.0f + 1e-5f);
  #pragma unroll
  for (int nt = 0; nt < 4; ++nt) {
    const int col = wc * 64 + nt * 16 + l15;
    const float gs = g[col] * rs;
    const float bb = bias[col];
    #pragma unroll
    for (int v = 0; v < 8; ++v) {
      int row = rowbase + v + hi * 8;
      float val = fmaxf(acc[nt][v] * gs + bb, 0.f);
      out[(size_t)row * HID + col] = (h16)val;
    }
  }
}

// ---------------- 6) final heads (1-wide, 7-wide) + decode + concat ----------------
__global__ __launch_bounds__(256) void head_final_kernel(
    const h16* __restrict__ c2, const h16* __restrict__ r2,
    const float* __restrict__ Wc3, const float* __restrict__ bc3,
    const float* __restrict__ Wr3, const float* __restrict__ br3,
    const float* __restrict__ rois, const float* __restrict__ roiscore,
    float* __restrict__ outp) {
  __shared__ float red[8 * 256];
  const int t = threadIdx.x;
  const int row = blockIdx.x;
  float cv = (float)c2[(size_t)row * HID + t];
  float rv = (float)r2[(size_t)row * HID + t];
  red[t] = cv * Wc3[t];
  #pragma unroll
  for (int o = 0; o < 7; ++o) red[(o + 1) * 256 + t] = rv * Wr3[o * 256 + t];
  __syncthreads();
  for (int s = 128; s > 0; s >>= 1) {
    if (t < s) {
      #pragma unroll
      for (int o = 0; o < 8; ++o) red[o * 256 + t] += red[o * 256 + t + s];
    }
    __syncthreads();
  }
  if (t == 0) {
    float cls = red[0] + bc3[0];
    float rg[7];
    #pragma unroll
    for (int o = 0; o < 7; ++o) rg[o] = red[(o + 1) * 256] + br3[o];
    const float* roi = rois + (size_t)row * 7;
    float dxa = roi[3], dya = roi[4], dza = roi[5], ra = roi[6];
    float diag = sqrtf(dxa * dxa + dya * dya);
    float xx = rg[0] * diag, yy = rg[1] * diag, zz = rg[2] * dza;
    float ddx = expf(rg[3]) * dxa, ddy = expf(rg[4]) * dya, ddz = expf(rg[5]) * dza;
    float head = rg[6] + ra;
    float cth = cosf(ra), sth = sinf(ra);
    float* o9 = outp + (size_t)row * 9;
    o9[0] = cls;
    o9[1] = xx * cth - yy * sth + roi[0];
    o9[2] = xx * sth + yy * cth + roi[1];
    o9[3] = zz + roi[2];
    o9[4] = ddx; o9[5] = ddy; o9[6] = ddz;
    o9[7] = head;
    o9[8] = roiscore[row];
  }
}

// ---------------- launch ----------------
static inline size_t alignup(size_t x) { return (x + 255) & ~(size_t)255; }

extern "C" void kernel_launch(void* const* d_in, const int* in_sizes, int n_in,
                              void* d_out, int out_size, void* d_ws, size_t ws_size,
                              hipStream_t stream) {
  (void)in_sizes; (void)n_in; (void)out_size; (void)ws_size;
  const float* rpn_box = (const float*)d_in[0];
  const float* rpn_cls = (const float*)d_in[1];
  const float* pooled  = (const float*)d_in[2];
  const float* Ws1 = (const float*)d_in[3];
  const float* g1  = (const float*)d_in[4];
  const float* b1  = (const float*)d_in[5];
  const float* Ws2 = (const float*)d_in[6];
  const float* g2  = (const float*)d_in[7];
  const float* b2  = (const float*)d_in[8];
  const float* Wc1 = (const float*)d_in[9];
  const float* gc1 = (const float*)d_in[10];
  const float* bc1 = (const float*)d_in[11];
  const float* Wc2 = (const float*)d_in[12];
  const float* gc2 = (const float*)d_in[13];
  const float* bc2 = (const float*)d_in[14];
  const float* Wc3 = (const float*)d_in[15];
  const float* bc3 = (const float*)d_in[16];
  const float* Wr1 = (const float*)d_in[17];
  const float* gr1 = (const float*)d_in[18];
  const float* br1 = (const float*)d_in[19];
  const float* Wr2 = (const float*)d_in[20];
  const float* gr2 = (const float*)d_in[21];
  const float* br2 = (const float*)d_in[22];
  const float* Wr3 = (const float*)d_in[23];
  const float* br3 = (const float*)d_in[24];

  char* w = (char*)d_ws;
  auto take = [&](size_t bytes) -> char* { char* p = w; w += alignup(bytes); return p; };
  float* ws_score    = (float*)take((size_t)B_ * A_ * 4);
  float* ws_topbox   = (float*)take((size_t)B_ * NPRE * 7 * 4);
  float* ws_topscore = (float*)take((size_t)B_ * NPRE * 4);
  float* ws_rois     = (float*)take((size_t)B_ * NPOST * 7 * 4);
  float* ws_roiscore = (float*)take((size_t)B_ * NPOST * 4);
  h16* w16_s1 = (h16*)take((size_t)HID * FEAT_ * 2);
  h16* w16_s2 = (h16*)take((size_t)HID * HID * 2);
  h16* w16_c1 = (h16*)take((size_t)HID * HID * 2);
  h16* w16_c2 = (h16*)take((size_t)HID * HID * 2);
  h16* w16_r1 = (h16*)take((size_t)HID * HID * 2);
  h16* w16_r2 = (h16*)take((size_t)HID * HID * 2);
  h16* act_x1 = (h16*)take((size_t)MROWS * HID * 2);
  h16* act_x2 = (h16*)take((size_t)MROWS * HID * 2);
  h16* act_c1 = (h16*)take((size_t)MROWS * HID * 2);
  h16* act_c2 = (h16*)take((size_t)MROWS * HID * 2);
  h16* act_r1 = (h16*)take((size_t)MROWS * HID * 2);
  h16* act_r2 = (h16*)take((size_t)MROWS * HID * 2);

  // proposal path
  score_kernel<<<(B_ * A_ + 255) / 256, 256, 0, stream>>>(rpn_cls, ws_score, B_ * A_);
  topk_kernel<<<B_, 1024, 0, stream>>>(ws_score, rpn_box, ws_topbox, ws_topscore);
  nms_select_kernel<<<B_, 1024, 0, stream>>>(ws_topbox, ws_topscore, ws_rois, ws_roiscore);

  // weight conversion (f32 -> f16, one-time)
  f32_to_f16_kernel<<<2048, 256, 0, stream>>>(Ws1, w16_s1, HID * FEAT_);
  f32_to_f16_kernel<<<128, 256, 0, stream>>>(Ws2, w16_s2, HID * HID);
  f32_to_f16_kernel<<<128, 256, 0, stream>>>(Wc1, w16_c1, HID * HID);
  f32_to_f16_kernel<<<128, 256, 0, stream>>>(Wc2, w16_c2, HID * HID);
  f32_to_f16_kernel<<<128, 256, 0, stream>>>(Wr1, w16_r1, HID * HID);
  f32_to_f16_kernel<<<128, 256, 0, stream>>>(Wr2, w16_r2, HID * HID);

  // WMMA MLP head
  const int gm = MROWS / 32;  // 64 workgroups
  gemm_bnrelu_kernel<true ><<<gm, 256, 0, stream>>>(pooled, w16_s1, g1, b1, act_x1, FEAT_);
  gemm_bnrelu_kernel<false><<<gm, 256, 0, stream>>>(act_x1, w16_s2, g2, b2, act_x2, HID);
  gemm_bnrelu_kernel<false><<<gm, 256, 0, stream>>>(act_x2, w16_c1, gc1, bc1, act_c1, HID);
  gemm_bnrelu_kernel<false><<<gm, 256, 0, stream>>>(act_c1, w16_c2, gc2, bc2, act_c2, HID);
  gemm_bnrelu_kernel<false><<<gm, 256, 0, stream>>>(act_x2, w16_r1, gr1, br1, act_r1, HID);
  gemm_bnrelu_kernel<false><<<gm, 256, 0, stream>>>(act_r1, w16_r2, gr2, br2, act_r2, HID);

  // final heads + box decode + concat
  head_final_kernel<<<MROWS, 256, 0, stream>>>(act_c2, act_r2, Wc3, bc3, Wr3, br3,
                                               ws_rois, ws_roiscore, (float*)d_out);
}
